// PairformerStack_7456063225922
// MI455X (gfx1250) — matrix operand order repaired and checked by reading, not compiled, not run on hardware
//
#include <hip/hip_runtime.h>
#include <hip/hip_bf16.h>
#include <math.h>

// ---------------------------------------------------------------------------
// CDNA5 / gfx1250 Pairformer stack.
// All GEMMs run through v_wmma_f32_16x16x32_bf16 (fp32 accumulate).
// Problem is L2-resident (~20MB state vs 192MB L2) -> matrix-throughput bound.
// Every GEMM is padded to 64x32x32 tile divisibility, so only two clean WMMA
// instantiations exist: vectorized-B (b128 loads both operands) and strided-B
// (hoisted loop-invariant offsets). No bounds checks, no exec-mask predication.
// Triangle einsums are pre-transposed to d-major (LDS-tiled transpose) so they
// run on the fully vectorized path. Activations use v_rcp_f32 (no IEEE div).
// ---------------------------------------------------------------------------

typedef __attribute__((ext_vector_type(16))) __bf16 v16bf;
typedef __attribute__((ext_vector_type(8)))  float  v8f;
typedef __attribute__((ext_vector_type(4)))  float  f4v;

#define ACT_NONE 0
#define ACT_SIG  1
#define ACT_SILU 2

namespace {
constexpr int       NSEQ = 192;
constexpr int       CZ   = 128, CS = 384;
constexpr int       HP   = 4,  DP = 32, HS = 16, DSd = 24, DPad = 32;
constexpr long long NN   = (long long)NSEQ * NSEQ;   // 36864
constexpr long long SZ   = NN * CZ;                  // 4718592 floats (18.9MB)
}

// ------------------------------ device helpers ------------------------------
__device__ __forceinline__ float wred_sum(float v) {
#pragma unroll
  for (int o = 16; o > 0; o >>= 1) v += __shfl_xor(v, o, 32);
  return v;
}
__device__ __forceinline__ float wred_max(float v) {
#pragma unroll
  for (int o = 16; o > 0; o >>= 1) v = fmaxf(v, __shfl_xor(v, o, 32));
  return v;
}
__device__ __forceinline__ float fast_sigmoid(float x) {
  // v_exp_f32 + v_rcp_f32: adequate for a bf16-precision pipeline
  return __builtin_amdgcn_rcpf(1.f + __expf(-x));
}

// ---------------------------------------------------------------------------
// Batched strided GEMM: C(m,n) = act(alpha * sum_k A(m,k)*B(k,n) + bias[n])
// Preconditions (enforced by host-side padding): M%64==0, N%32==0, K%32==0,
// A K-contiguous (sAk==1). Batch bz = bo*bh + bi with (outer,inner) strides.
// Block = 256 threads = 8 waves (4 along M x 2 along N); each wave owns a
// 16x16 C tile, K consumed 32/step by v_wmma_f32_16x16x32_bf16.
// Fragment layouts used here (wave32; A/B contiguous k, C per CDNA5 ISA 7.12.2):
//   A 16x32 bf16: lanes 0-15 rows M, elems K0..15; lanes 16-31 K16..31
//   B 32x16 bf16: lane&15 = N col; lanes 0-15 K0..15, lanes 16-31 K16..31
//   C f32: VGPR r -> M = r (lanes 0-15) / r+8 (lanes 16-31), N = lane&15
// BV: B K-contiguous -> b128 vector loads; else bumped pointer + hoisted
//     loop-invariant per-element offsets (e*sBk).
// ---------------------------------------------------------------------------
template <bool BV>
__global__ void gemm_wmma_kernel(const float* __restrict__ A,
                                 const float* __restrict__ B,
                                 const float* __restrict__ bias,
                                 float* __restrict__ C,
                                 int K, int bh,
                                 long long sAm, long long bsA1, long long bsA2,
                                 long long sBk, long long sBn, long long bsB1, long long bsB2,
                                 long long sCm, long long sCn, long long bsC1, long long bsC2,
                                 float alpha, int act)
{
  const int bz = blockIdx.z;
  const int bo = bz / bh, bi = bz % bh;
  A += bo * bsA1 + bi * bsA2;
  B += bo * bsB1 + bi * bsB2;
  C += bo * bsC1 + bi * bsC2;

  const int wave = threadIdx.x >> 5;
  const int lane = threadIdx.x & 31;
  const int wm = wave & 3;          // 4 waves along M
  const int wn = wave >> 2;         // 2 waves along N
  const int tm = blockIdx.y * 64 + wm * 16;
  const int tn = blockIdx.x * 32 + wn * 16;

  const int mrow  = tm + (lane & 15);
  const int ncol  = tn + (lane & 15);
  const int aksel = (lane >> 4) * 16;          // A: upper lanes hold K+16 half
  const int bksel = (lane >> 4) * 16;          // B: upper lanes hold K+16 half

  const float* aPtr = A + (long long)mrow * sAm + aksel;
  const float* bPtr = B + (long long)ncol * sBn + (BV ? (long long)bksel
                                                      : (long long)bksel * sBk);
  v8f acc = {};

  for (int k0 = 0; k0 < K; k0 += 32) {
    v16bf af, bf;
    const f4v a0 = *reinterpret_cast<const f4v*>(aPtr);
    const f4v a1 = *reinterpret_cast<const f4v*>(aPtr + 4);
    const f4v a2 = *reinterpret_cast<const f4v*>(aPtr + 8);
    const f4v a3 = *reinterpret_cast<const f4v*>(aPtr + 12);
#pragma unroll
    for (int j = 0; j < 4; ++j) {
      af[j]      = (__bf16)a0[j];
      af[j + 4]  = (__bf16)a1[j];
      af[j + 8]  = (__bf16)a2[j];
      af[j + 12] = (__bf16)a3[j];
    }
    if constexpr (BV) {
      const f4v b0 = *reinterpret_cast<const f4v*>(bPtr);
      const f4v b1 = *reinterpret_cast<const f4v*>(bPtr + 4);
      const f4v b2 = *reinterpret_cast<const f4v*>(bPtr + 8);
      const f4v b3 = *reinterpret_cast<const f4v*>(bPtr + 12);
#pragma unroll
      for (int j = 0; j < 4; ++j) {
        bf[j]      = (__bf16)b0[j];
        bf[j + 4]  = (__bf16)b1[j];
        bf[j + 8]  = (__bf16)b2[j];
        bf[j + 12] = (__bf16)b3[j];
      }
      bPtr += 32;
    } else {
      // offsets e*sBk are loop-invariant -> hoisted out of the k-loop
#pragma unroll
      for (int e = 0; e < 16; ++e) bf[e] = (__bf16)bPtr[(long long)e * sBk];
      bPtr += 32 * sBk;
    }
    aPtr += 32;
    acc = __builtin_amdgcn_wmma_f32_16x16x32_bf16(
        false, af, false, bf, (short)0, acc, false, false);
  }

#pragma unroll
  for (int r = 0; r < 8; ++r) {
    const int m = tm + r + ((lane >> 4) << 3);
    float v = acc[r] * alpha;
    if (bias) v += bias[ncol];
    if (act == ACT_SIG)       v = fast_sigmoid(v);
    else if (act == ACT_SILU) v = v * fast_sigmoid(v);
    C[(long long)m * sCm + (long long)ncol * sCn] = v;
  }
}

// ----------------- LDS-tiled transposes for d-major einsum ------------------
// to_dmajor:  dst[d][x] = src[perm(x)*128 + d],  x in [0, 36864), d in [0,128)
//   swap=0: perm(x) = x                     (a[i,k,d] -> aT[d][i*192+k])
//   swap=1: perm(x) = (x%192)*192 + x/192   (a[k,i,d] -> aT[d][i*192+k])
__global__ void to_dmajor_kernel(float* __restrict__ dst, const float* __restrict__ src,
                                 int swap)
{
  __shared__ float tile[32][33];
  const int x0 = blockIdx.x * 32;
  const int d0 = blockIdx.y * 32;
  const int tx = threadIdx.x & 31, ty = threadIdx.x >> 5;
#pragma unroll
  for (int yy = ty; yy < 32; yy += 8) {
    const int x = x0 + yy;
    const int xi = swap ? ((x % NSEQ) * NSEQ + x / NSEQ) : x;
    tile[yy][tx] = src[(long long)xi * CZ + d0 + tx];
  }
  __syncthreads();
#pragma unroll
  for (int yy = ty; yy < 32; yy += 8)
    dst[(long long)(d0 + yy) * NN + x0 + tx] = tile[tx][yy];
}

// from_dmajor: dst[x*128 + d] = src[d][x]
__global__ void from_dmajor_kernel(float* __restrict__ dst, const float* __restrict__ src)
{
  __shared__ float tile[32][33];
  const int x0 = blockIdx.x * 32;
  const int d0 = blockIdx.y * 32;
  const int tx = threadIdx.x & 31, ty = threadIdx.x >> 5;
#pragma unroll
  for (int yy = ty; yy < 32; yy += 8)
    tile[yy][tx] = src[(long long)(d0 + yy) * NN + x0 + tx];
  __syncthreads();
#pragma unroll
  for (int yy = ty; yy < 32; yy += 8)
    dst[(long long)(x0 + yy) * CZ + d0 + tx] = tile[tx][yy];
}

// -------------------- pad/crop columns (zero-fill padding) ------------------
// dst[r*dstC + c] = (c < copyC) ? src[r*srcC + c] : 0
__global__ void copy_cols_kernel(float* __restrict__ dst, const float* __restrict__ src,
                                 int rows, int dstC, int srcC, int copyC)
{
  long long i = (long long)blockIdx.x * blockDim.x + threadIdx.x;
  const long long n = (long long)rows * dstC;
  const long long stp = (long long)gridDim.x * blockDim.x;
  for (; i < n; i += stp) {
    const int c = (int)(i % dstC);
    const long long r = i / dstC;
    dst[i] = (c < copyC) ? src[r * srcC + c] : 0.f;
  }
}

// ------------------------------ LayerNorm -----------------------------------
__global__ void ln_kernel(const float* __restrict__ x, const float* __restrict__ g,
                          const float* __restrict__ b, float* __restrict__ y,
                          int rows, int C)
{
  const int wave = threadIdx.x >> 5, lane = threadIdx.x & 31;
  const int row = blockIdx.x * (blockDim.x >> 5) + wave;
  if (row >= rows) return;
  const float* xr = x + (long long)row * C;
  float* yr = y + (long long)row * C;
  float s = 0.f;
  for (int c = lane; c < C; c += 32) s += xr[c];
  const float mean = wred_sum(s) / (float)C;
  float v = 0.f;
  for (int c = lane; c < C; c += 32) { float d = xr[c] - mean; v += d * d; }
  const float inv = rsqrtf(wred_sum(v) / (float)C + 1e-5f);
  for (int c = lane; c < C; c += 32) yr[c] = (xr[c] - mean) * inv * g[c] + b[c];
}

// -------------------- softmax over cols with strided bias -------------------
// row index = b*R + r ; val = s[c]*scale + bias[r*br + c*bc + (b%H)]
__global__ void softmax_bias_kernel(float* __restrict__ sc, const float* __restrict__ bias,
                                    int batch, int R, int C, int H,
                                    long long br, long long bc, float scale)
{
  const int wave = threadIdx.x >> 5, lane = threadIdx.x & 31;
  const int row = blockIdx.x * (blockDim.x >> 5) + wave;
  if (row >= batch * R) return;
  const int b = row / R, r = row - b * R;
  const int h = b % H;
  float* s = sc + (long long)row * C;
  const float* bp = bias + (long long)r * br + h;
  float mx = -3.0e38f;
  for (int c = lane; c < C; c += 32) {
    float v = s[c] * scale + bp[(long long)c * bc];
    s[c] = v;
    mx = fmaxf(mx, v);
  }
  mx = wred_max(mx);
  float sum = 0.f;
  for (int c = lane; c < C; c += 32) { float e = __expf(s[c] - mx); s[c] = e; sum += e; }
  sum = wred_sum(sum);
  const float inv = __builtin_amdgcn_rcpf(sum);
  for (int c = lane; c < C; c += 32) s[c] *= inv;
}

// ------------------------------ elementwise ---------------------------------
__global__ void ew_mul_kernel(float* __restrict__ o, const float* __restrict__ a,
                              const float* __restrict__ b, long long n)
{
  long long i = (long long)blockIdx.x * blockDim.x + threadIdx.x;
  const long long st = (long long)gridDim.x * blockDim.x;
  for (; i < n; i += st) o[i] = a[i] * b[i];
}
__global__ void ew_mul_add_kernel(float* __restrict__ d, const float* __restrict__ g,
                                  const float* __restrict__ x, long long n)
{
  long long i = (long long)blockIdx.x * blockDim.x + threadIdx.x;
  const long long st = (long long)gridDim.x * blockDim.x;
  for (; i < n; i += st) d[i] += g[i] * x[i];
}
__global__ void ew_add_kernel(float* __restrict__ d, const float* __restrict__ s, long long n)
{
  long long i = (long long)blockIdx.x * blockDim.x + threadIdx.x;
  const long long st = (long long)gridDim.x * blockDim.x;
  for (; i < n; i += st) d[i] += s[i];
}

// ------------------------------ host wrappers -------------------------------
static void gemm(hipStream_t st, const float* A, const float* B, const float* bias, float* C,
                 int M, int N, int K, int batch, int bh,
                 long long sAm, long long bsA1, long long bsA2,
                 long long sBk, long long sBn, long long bsB1, long long bsB2,
                 long long sCm, long long sCn, long long bsC1, long long bsC2,
                 float alpha, int act)
{
  dim3 grid(N / 32, M / 64, batch);   // M%64==0, N%32==0, K%32==0 guaranteed
#define GEMM_ARGS A, B, bias, C, K, bh, sAm, bsA1, bsA2, \
                  sBk, sBn, bsB1, bsB2, sCm, sCn, bsC1, bsC2, alpha, act
  if (sBk == 1) gemm_wmma_kernel<true ><<<grid, 256, 0, st>>>(GEMM_ARGS);
  else          gemm_wmma_kernel<false><<<grid, 256, 0, st>>>(GEMM_ARGS);
#undef GEMM_ARGS
}
static void gemm_rm(hipStream_t st, const float* A, const float* B, const float* bias,
                    float* C, int M, int N, int K, int act)
{
  gemm(st, A, B, bias, C, M, N, K, 1, 1,
       K, 0, 0,  N, 1, 0, 0,  N, 1, 0, 0, 1.f, act);
}
static void to_dmajor(hipStream_t st, float* dst, const float* src, int swap)
{
  dim3 grid((int)(NN / 32), CZ / 32, 1);
  to_dmajor_kernel<<<grid, 256, 0, st>>>(dst, src, swap);
}
static void from_dmajor(hipStream_t st, float* dst, const float* src)
{
  dim3 grid((int)(NN / 32), CZ / 32, 1);
  from_dmajor_kernel<<<grid, 256, 0, st>>>(dst, src);
}
static void copy_cols(hipStream_t st, float* dst, int dstC, const float* src, int srcC,
                      int rows, int copyC)
{
  long long n = (long long)rows * dstC;
  copy_cols_kernel<<<(int)((n + 255) / 256), 256, 0, st>>>(dst, src, rows, dstC, srcC, copyC);
}
static void ln(hipStream_t st, float* y, const float* x, const float* g, const float* b,
               int rows, int C)
{
  ln_kernel<<<(rows + 7) / 8, 256, 0, st>>>(x, g, b, y, rows, C);
}
static void softmax(hipStream_t st, float* sc, const float* bias, int batch, int R, int C,
                    int H, long long br, long long bc, float scale)
{
  int rows = batch * R;
  softmax_bias_kernel<<<(rows + 7) / 8, 256, 0, st>>>(sc, bias, batch, R, C, H, br, bc, scale);
}
static void ew_mul(hipStream_t st, float* o, const float* a, const float* b, long long n)
{
  ew_mul_kernel<<<(int)((n + 255) / 256), 256, 0, st>>>(o, a, b, n);
}
static void ew_mul_add(hipStream_t st, float* d, const float* g, const float* x, long long n)
{
  ew_mul_add_kernel<<<(int)((n + 255) / 256), 256, 0, st>>>(d, g, x, n);
}
static void ew_add(hipStream_t st, float* d, const float* s, long long n)
{
  ew_add_kernel<<<(int)((n + 255) / 256), 256, 0, st>>>(d, s, n);
}

// ------------------------------ module functions ----------------------------
static void tri_mul(hipStream_t st, float* z,
                    const float* lnin_g, const float* lnin_b,
                    const float* Wa1, const float* Wa2,
                    const float* Wb1, const float* Wb2,
                    const float* Wg, const float* Wout,
                    const float* lnout_g, const float* lnout_b,
                    bool outgoing,
                    float* B0, float* B1, float* B2, float* B3, float* B4,
                    float* B5, float* B6)
{
  ln(st, B0, z, lnin_g, lnin_b, (int)NN, CZ);                       // zl
  gemm_rm(st, B0, Wa1, nullptr, B1, (int)NN, CZ, CZ, ACT_SIG);
  gemm_rm(st, B0, Wa2, nullptr, B2, (int)NN, CZ, CZ, ACT_NONE);
  ew_mul(st, B1, B1, B2, SZ);                                       // a
  gemm_rm(st, B0, Wb1, nullptr, B2, (int)NN, CZ, CZ, ACT_SIG);
  gemm_rm(st, B0, Wb2, nullptr, B3, (int)NN, CZ, CZ, ACT_NONE);
  ew_mul(st, B2, B2, B3, SZ);                                       // b
  gemm_rm(st, B0, Wg, nullptr, B3, (int)NN, CZ, CZ, ACT_SIG);       // g
  // d-major einsum: aT[d][i*192+k], bT[d][j*192+k]; S_d[d][i][j] = aT @ bT^T
  const int swap = outgoing ? 0 : 1;
  to_dmajor(st, B4, B1, swap);                                      // aT
  to_dmajor(st, B5, B2, swap);                                      // bT
  gemm(st, B4, B5, nullptr, B6, NSEQ, NSEQ, NSEQ, CZ, 1,
       NSEQ, NN, 0,    1, NSEQ, NN, 0,    NSEQ, 1, NN, 0, 1.f, ACT_NONE);
  from_dmajor(st, B1, B6);                                          // S[i,j,d]
  ln(st, B0, B1, lnout_g, lnout_b, (int)NN, CZ);
  gemm_rm(st, B0, Wout, nullptr, B1, (int)NN, CZ, CZ, ACT_NONE);
  ew_mul_add(st, z, B3, B1, SZ);                                    // z += g * out
}

static void tri_attn(hipStream_t st, float* z,
                     const float* ln_g, const float* ln_b,
                     const float* Wq, const float* Wk, const float* Wv,
                     const float* Wb, const float* Wg, const float* Wout,
                     bool starting,
                     float* B0, float* B1, float* B2, float* B3, float* B4,
                     float* B5, float* B6, float* bij, float* wbPad)
{
  ln(st, B0, z, ln_g, ln_b, (int)NN, CZ);
  gemm_rm(st, B0, Wq, nullptr, B1, (int)NN, CZ, CZ, ACT_NONE);      // q
  gemm_rm(st, B0, Wk, nullptr, B2, (int)NN, CZ, CZ, ACT_NONE);      // k
  gemm_rm(st, B0, Wv, nullptr, B3, (int)NN, CZ, CZ, ACT_NONE);      // v
  gemm_rm(st, B0, Wg, nullptr, B4, (int)NN, CZ, CZ, ACT_SIG);       // gate
  copy_cols(st, wbPad, 32, Wb, HP, CZ, HP);                         // Wb 128x4 -> 128x32
  gemm_rm(st, B0, wbPad, nullptr, bij, (int)NN, 32, CZ, ACT_NONE);  // bij[x][32]
  const float scale = 0.17677669529663687f;                         // 1/sqrt(32)
  const long long R = (long long)NSEQ * CZ;                         // row stride in q/k/v
  const long long BR = (long long)NSEQ * 32;                        // bij row stride
  for (int c0 = 0; c0 < NSEQ; c0 += 32) {                           // 32 outer idx/chunk
    if (starting) {
      // scores[(ii,h)][j][k] = sum_d q[ii,j,h,d]*k[ii,k,h,d]
      gemm(st, B1 + (long long)c0 * R, B2 + (long long)c0 * R, nullptr, B5,
           NSEQ, NSEQ, DP, 32 * HP, HP,
           CZ, R, DP,    1, CZ, R, DP,
           NSEQ, 1, (long long)HP * NN, NN, 1.f, ACT_NONE);
      softmax(st, B5, bij, 32 * HP, NSEQ, NSEQ, HP, BR, 32, scale); // + bij[j,k,h]
      // o[ii,j,h,d] = sum_k A[j][k]*v[ii,k,h,d]
      gemm(st, B5, B3 + (long long)c0 * R, nullptr, B6 + (long long)c0 * R,
           NSEQ, DP, NSEQ, 32 * HP, HP,
           NSEQ, (long long)HP * NN, NN,
           CZ, 1, R, DP,    CZ, 1, R, DP, 1.f, ACT_NONE);
    } else {
      // scores[(jj,h)][i][k] = sum_d q[i,jj,h,d]*k[k,jj,h,d]
      gemm(st, B1 + (long long)c0 * CZ, B2 + (long long)c0 * CZ, nullptr, B5,
           NSEQ, NSEQ, DP, 32 * HP, HP,
           R, CZ, DP,    1, R, CZ, DP,
           NSEQ, 1, (long long)HP * NN, NN, 1.f, ACT_NONE);
      softmax(st, B5, bij, 32 * HP, NSEQ, NSEQ, HP, 32, BR, scale); // + bij[k,i,h]
      // o[i,jj,h,d] = sum_k A[i][k]*v[k,jj,h,d]
      gemm(st, B5, B3 + (long long)c0 * CZ, nullptr, B6 + (long long)c0 * CZ,
           NSEQ, DP, NSEQ, 32 * HP, HP,
           NSEQ, (long long)HP * NN, NN,
           R, 1, CZ, DP,    R, 1, CZ, DP, 1.f, ACT_NONE);
    }
  }
  ew_mul(st, B1, B6, B4, SZ);                                       // g * o
  gemm_rm(st, B1, Wout, nullptr, B2, (int)NN, CZ, CZ, ACT_NONE);
  ew_add(st, z, B2, SZ);
}

static void transition(hipStream_t st, float* x, int rows, int C, int hidden,
                       const float* ln_g, const float* ln_b,
                       const float* W1, const float* W2, const float* Wout,
                       float* Bln, float* T1, float* T2, float* Tout, int rowchunk)
{
  ln(st, Bln, x, ln_g, ln_b, rows, C);
  for (int r0 = 0; r0 < rows; r0 += rowchunk) {
    const int rc = (rows - r0 < rowchunk) ? (rows - r0) : rowchunk;
    const float* xl = Bln + (long long)r0 * C;
    gemm_rm(st, xl, W1, nullptr, T1, rc, hidden, C, ACT_SILU);
    gemm_rm(st, xl, W2, nullptr, T2, rc, hidden, C, ACT_NONE);
    ew_mul(st, T1, T1, T2, (long long)rc * hidden);
    gemm_rm(st, T1, Wout, nullptr, Tout, rc, C, hidden, ACT_NONE);
    ew_add(st, x + (long long)r0 * C, Tout, (long long)rc * C);
  }
}

static void apb_fn(hipStream_t st, float* s, const float* z,
                   const float* lnA_g, const float* lnA_b,
                   const float* Wq, const float* bq,
                   const float* Wk, const float* Wv,
                   const float* lnB_g, const float* lnB_b, const float* Wb,
                   const float* Wg, const float* Wout,
                   float* Bz, float* sa, float* sq, float* sk, float* sv,
                   float* sg, float* so, float* stmp, float* bij, float* ssc,
                   float* qp, float* kp, float* vp, float* op, float* wbPad)
{
  ln(st, sa, s, lnA_g, lnA_b, NSEQ, CS);
  gemm_rm(st, sa, Wq, bq,      sq, NSEQ, CS, CS, ACT_NONE);
  gemm_rm(st, sa, Wk, nullptr, sk, NSEQ, CS, CS, ACT_NONE);
  gemm_rm(st, sa, Wv, nullptr, sv, NSEQ, CS, CS, ACT_NONE);
  gemm_rm(st, sa, Wg, nullptr, sg, NSEQ, CS, CS, ACT_SIG);
  ln(st, Bz, z, lnB_g, lnB_b, (int)NN, CZ);
  copy_cols(st, wbPad, 32, Wb, HS, CZ, HS);                         // Wb 128x16 -> 128x32
  gemm_rm(st, Bz, wbPad, nullptr, bij, (int)NN, 32, CZ, ACT_NONE);  // bij[x][32]
  // zero-pad heads 24 -> 32 so all GEMMs are tile-divisible (zeros are exact)
  copy_cols(st, qp, DPad, sq, DSd, NSEQ * HS, DSd);
  copy_cols(st, kp, DPad, sk, DSd, NSEQ * HS, DSd);
  copy_cols(st, vp, DPad, sv, DSd, NSEQ * HS, DSd);
  const long long PR = (long long)HS * DPad;                        // 512
  // scores[h][i][j] = sum_d qp[i,h,d]*kp[j,h,d]   (K=32, d>=24 contribute 0)
  gemm(st, qp, kp, nullptr, ssc, NSEQ, NSEQ, DPad, HS, HS,
       PR, 0, DPad,    1, PR, 0, DPad,
       NSEQ, 1, 0, NN, 1.f, ACT_NONE);
  softmax(st, ssc, bij, HS, NSEQ, NSEQ, HS,
          (long long)NSEQ * 32, 32, 0.20412414523193154f);          // 1/sqrt(24)
  // op[i,h,d] = sum_j A[h][i][j]*vp[j,h,d]
  gemm(st, ssc, vp, nullptr, op, NSEQ, DPad, NSEQ, HS, HS,
       NSEQ, 0, NN,    PR, 1, 0, DPad,    PR, 1, 0, DPad, 1.f, ACT_NONE);
  copy_cols(st, so, DSd, op, DPad, NSEQ * HS, DSd);                 // crop 32 -> 24
  ew_mul(st, so, so, sg, (long long)NSEQ * CS);
  gemm_rm(st, so, Wout, nullptr, stmp, NSEQ, CS, CS, ACT_NONE);
  ew_add(st, s, stmp, (long long)NSEQ * CS);
}

// ------------------------------ entry point ---------------------------------
extern "C" void kernel_launch(void* const* d_in, const int* in_sizes, int n_in,
                              void* d_out, int out_size, void* d_ws, size_t ws_size,
                              hipStream_t stream)
{
  (void)in_sizes; (void)n_in; (void)out_size; (void)ws_size;

  const float* s_in = (const float*)d_in[0];
  const float* z_in = (const float*)d_in[1];
  float* s = (float*)d_out;                      // 192*384
  float* z = s + (size_t)NSEQ * CS;              // 192*192*128
  hipMemcpyAsync(s, s_in, sizeof(float) * (size_t)NSEQ * CS, hipMemcpyDeviceToDevice, stream);
  hipMemcpyAsync(z, z_in, sizeof(float) * (size_t)SZ, hipMemcpyDeviceToDevice, stream);

  // workspace: 7 z-sized buffers + small pool (~145MB total)
  float* ws = (float*)d_ws;
  float* B0 = ws + 0 * SZ;
  float* B1 = ws + 1 * SZ;
  float* B2 = ws + 2 * SZ;
  float* B3 = ws + 3 * SZ;
  float* B4 = ws + 4 * SZ;
  float* B5 = ws + 5 * SZ;   // attention score chunk / einsum bT
  float* B6 = ws + 6 * SZ;   // attention out / einsum S_d
  float* SM = ws + 7 * SZ;
  float* sm_a   = SM;                   // 73728
  float* sm_q   = sm_a   + 73728;
  float* sm_k   = sm_q   + 73728;
  float* sm_v   = sm_k   + 73728;
  float* sm_g   = sm_v   + 73728;
  float* sm_o   = sm_g   + 73728;
  float* sm_t   = sm_o   + 73728;
  float* sm_bij = sm_t   + 73728;       // NN*32 = 1179648 (tri + apb padded bias)
  float* sm_sc  = sm_bij + 1179648;     // 589824
  float* sm_t1  = sm_sc  + 589824;      // 294912
  float* sm_t2  = sm_t1  + 294912;      // 294912
  float* sm_qp  = sm_t2  + 294912;      // 98304 (192*16*32)
  float* sm_kp  = sm_qp  + 98304;
  float* sm_vp  = sm_kp  + 98304;
  float* sm_op  = sm_vp  + 98304;
  float* sm_wb  = sm_op  + 98304;       // 4096 (128*32 padded Wb)

  for (int blk = 0; blk < 2; ++blk) {
    const int base = 2 + blk * 57;     // pytree leaves: sorted dict keys
    auto P = [&](int i) { return (const float*)d_in[base + i]; };
    // leaf offsets: apb 0..10 {Wb,Wg,Wk,Wout,Wq,Wv,bq,lnA_b,lnA_g,lnB_b,lnB_g}
    //               tae 11..18 / tas 19..26 {Wb,Wg,Wk,Wout,Wq,Wv,ln_b,ln_g}
    //               tmi 27..36 / tmo 37..46 {Wa1,Wa2,Wb1,Wb2,Wg,Wout,ln_in_b,ln_in_g,ln_out_b,ln_out_g}
    //               trs 47..51 / trz 52..56 {W1,W2,Wout,ln_b,ln_g}
    tri_mul(stream, z, P(44), P(43), P(37), P(38), P(39), P(40), P(41), P(42),
            P(46), P(45), true,  B0, B1, B2, B3, B4, B5, B6);                // tmo
    tri_mul(stream, z, P(34), P(33), P(27), P(28), P(29), P(30), P(31), P(32),
            P(36), P(35), false, B0, B1, B2, B3, B4, B5, B6);                // tmi
    tri_attn(stream, z, P(26), P(25), P(23), P(21), P(24), P(19), P(20), P(22),
             true,  B0, B1, B2, B3, B4, B5, B6, sm_bij, sm_wb);              // tas
    tri_attn(stream, z, P(18), P(17), P(15), P(13), P(16), P(11), P(12), P(14),
             false, B0, B1, B2, B3, B4, B5, B6, sm_bij, sm_wb);              // tae
    transition(stream, z, (int)NN, CZ, 4 * CZ, P(56), P(55), P(52), P(53), P(54),
               B0, B1, B2, B3, 9216);                                        // trz
    apb_fn(stream, s, z, P(8), P(7), P(4), P(6), P(2), P(5), P(10), P(9),
           P(0), P(1), P(3), B0, sm_a, sm_q, sm_k, sm_v, sm_g, sm_o, sm_t,
           sm_bij, sm_sc, sm_qp, sm_kp, sm_vp, sm_op, sm_wb);                // apb
    transition(stream, s, NSEQ, CS, 4 * CS, P(51), P(50), P(47), P(48), P(49),
               sm_a, sm_t1, sm_t2, sm_t, NSEQ);                              // trs
  }
}
